// SliceEncoder_34437047779370
// MI455X (gfx1250) — matrix operand order repaired and checked by reading, not compiled, not run on hardware
//
#include <hip/hip_runtime.h>
#include <stdint.h>

// ---------------------------------------------------------------------------
// SliceEncoder for MI455X (gfx1250, wave32).
// Heavy op: (25088 x 768) x (768 x 768) patch-embed GEMM -> bf16 WMMA
// (v_wmma_f32_16x16x32_bf16), f32 accumulation.
// Block tile: 32 rows x full E. Each wave owns 2 M-tiles and, in two
// passes, 2x3 e-tiles: per K-step 4 ds_load_b128 + 6 global_load_b128
// feed 6 independent WMMAs (~1 VMEM instr per WMMA; B reused across the
// two M-tiles, A-frag reused across the 3 e-tiles).
// ---------------------------------------------------------------------------

typedef __attribute__((ext_vector_type(16))) __bf16 v16bf;
typedef __attribute__((ext_vector_type(8)))  float  v8f;

#define BATCH   4
#define N_ALL   128
#define NKEEP   64
#define M_ALL   196
#define MKEEP   98
#define E_DIM   768
#define K_DIM   768          // C*PW*PH = 3*16*16
#define ROWS    (BATCH*NKEEP*MKEEP)      // 25088
#define X_SIZE  (ROWS*E_DIM)             // 19,267,584 floats
#define IMG_SLICE   (224*224)            // 50176
#define IMG_CSTRIDE (N_ALL*IMG_SLICE)    // channel stride in img: 6,422,528
#define A_STRIDE 776                     // padded LDS row stride (bf16 elems)
#define MT      2                        // M-tiles (16 rows each) per block
#define EJ      3                        // e-tiles per wave per half

static __device__ __forceinline__ unsigned short f2bf(float f) {
    // round-to-nearest-even f32 -> bf16
    unsigned u = __float_as_uint(f);
    unsigned r = u + 0x7FFFu + ((u >> 16) & 1u);
    return (unsigned short)(r >> 16);
}

// ---------------------------------------------------------------------------
// Kernel 1: outer argsort (B=4 rows of 128). Rank by counting (stable).
// ---------------------------------------------------------------------------
__global__ __launch_bounds__(128) void k_outer(const float* __restrict__ noise,
                                               int* __restrict__ wkeep,
                                               float* __restrict__ out_keep,
                                               float* __restrict__ out_drop,
                                               float* __restrict__ out_restore) {
    int b = blockIdx.x;
    int t = threadIdx.x;
    __shared__ float s[N_ALL];
    s[t] = noise[b * N_ALL + t];
    __syncthreads();
    float v = s[t];
    int rank = 0;
    #pragma unroll 8
    for (int i = 0; i < N_ALL; ++i) {
        float u = s[i];
        rank += (u < v) || (u == v && i < t);
    }
    if (rank < NKEEP) {
        wkeep[b * NKEEP + rank]    = t;
        out_keep[b * NKEEP + rank] = (float)t;
    } else {
        out_drop[b * NKEEP + (rank - NKEEP)] = (float)t;
    }
    out_restore[b * N_ALL + t] = (float)rank;
}

// ---------------------------------------------------------------------------
// Kernel 2: inner argsort per (b,n): 256 rows of 196, keep first 98 ranks.
// ---------------------------------------------------------------------------
__global__ __launch_bounds__(256) void k_inner(const float* __restrict__ noise,
                                               int* __restrict__ wkeepp) {
    int bn = blockIdx.x;                 // 0..255
    int t  = threadIdx.x;
    __shared__ float s[M_ALL];
    if (t < M_ALL) s[t] = noise[bn * M_ALL + t];
    __syncthreads();
    if (t < M_ALL) {
        float v = s[t];
        int rank = 0;
        for (int i = 0; i < M_ALL; ++i) {
            float u = s[i];
            rank += (u < v) || (u == v && i < t);
        }
        if (rank < MKEEP) wkeepp[bn * MKEEP + rank] = t;
    }
}

// ---------------------------------------------------------------------------
// Kernel 3: proj_w f32 (E,K) -> bf16 (E,K) row-major in workspace.
// ---------------------------------------------------------------------------
__global__ __launch_bounds__(256) void k_wconv(const float* __restrict__ w,
                                               unsigned short* __restrict__ wbf) {
    int i = blockIdx.x * 1024 + threadIdx.x;
    #pragma unroll
    for (int j = 0; j < 4; ++j) {
        int idx = i + j * 256;
        if (idx < E_DIM * K_DIM) wbf[idx] = f2bf(w[idx]);
    }
}

// ---------------------------------------------------------------------------
// Kernel 4: main GEMM. One block = 32 output rows x full E (48 e-tiles).
// ---------------------------------------------------------------------------
__global__ __launch_bounds__(256) void k_gemm(const float* __restrict__ img,
                                              const int* __restrict__ wkeep,
                                              const int* __restrict__ wkeepp,
                                              const unsigned short* __restrict__ wbf,
                                              const float* __restrict__ proj_b,
                                              const float* __restrict__ inner_pos,
                                              float* __restrict__ out) {
    __shared__ unsigned short As[32 * A_STRIDE];  // 32 rows x 768 bf16, +8 pad/row
    __shared__ int s_mid[32];
    __shared__ int s_base[32];

    const int tid = threadIdx.x;
    const int r0  = blockIdx.x * 32;             // global row base (784 blocks)

    // Resolve gather indices for the 32 rows of this tile.
    if (tid < 32) {
        int g   = r0 + tid;
        int b   = g / (NKEEP * MKEEP);
        int rem = g - b * (NKEEP * MKEEP);
        int n   = rem / MKEEP;
        int mi  = rem - n * MKEEP;
        int m   = wkeepp[(b * NKEEP + n) * MKEEP + mi];
        int nid = wkeep[b * NKEEP + n];
        int wb  = m / 14;
        int hb  = m - wb * 14;
        s_mid[tid]  = m;
        // img layout (B,C,N,W,H); base with c=0; element (c,pw,ph) adds
        // c*IMG_CSTRIDE + pw*14*224 + ph*14.
        s_base[tid] = (b * 3 * N_ALL + nid) * IMG_SLICE + wb * 224 + hb;
    }
    __syncthreads();

    // Cooperative gather: 32*768 = 24576 f32 -> bf16 in LDS (96 per thread).
    // Pixels are consumed exactly once -> non-temporal loads keep L2 for B.
    #pragma unroll 4
    for (int j = 0; j < 96; ++j) {
        int idx = j * 256 + tid;
        int row = idx / K_DIM;
        int k   = idx - row * K_DIM;
        int c   = k >> 8;
        int pw  = (k >> 4) & 15;
        int ph  = k & 15;
        float v = __builtin_nontemporal_load(
            img + s_base[row] + c * IMG_CSTRIDE + pw * (14 * 224) + ph * 14);
        As[row * A_STRIDE + k] = f2bf(v);
    }
    __syncthreads();

    const int lane = tid & 31;
    const int wave = tid >> 5;
    const int nloc = lane & 15;      // A: row m / B: col n / C: col n
    const int h    = lane >> 4;      // lane half

    // A fragment bases for the two M-tiles (ISA 16-bit A layout: K-local 8h).
    const unsigned short* aRow0 = &As[nloc * A_STRIDE + 8 * h];
    const unsigned short* aRow1 = &As[(16 + nloc) * A_STRIDE + 8 * h];

    // Two halves of 3 e-tiles each; within a half the 3 e-tiles are a
    // constant 128 cols = 196,608 B apart (24-bit immediate friendly).
    for (int half = 0; half < 2; ++half) {
        const unsigned short* bPtr = wbf
            + (size_t)(half * 384 + wave * 16 + nloc) * K_DIM + 8 * h;

        v8f acc[MT][EJ];
        #pragma unroll
        for (int mt = 0; mt < MT; ++mt)
            #pragma unroll
            for (int j = 0; j < EJ; ++j)
                acc[mt][j] = (v8f){0.f, 0.f, 0.f, 0.f, 0.f, 0.f, 0.f, 0.f};

        #pragma unroll 2
        for (int k0 = 0; k0 < K_DIM; k0 += 32) {
            v16bf aF[MT];
            ((uint4*)&aF[0])[0] = *(const uint4*)(aRow0 + k0);      // K=k0+8h..
            ((uint4*)&aF[0])[1] = *(const uint4*)(aRow0 + k0 + 16); // K=k0+16+8h..
            ((uint4*)&aF[1])[0] = *(const uint4*)(aRow1 + k0);
            ((uint4*)&aF[1])[1] = *(const uint4*)(aRow1 + k0 + 16);

            v16bf bF[EJ];
            #pragma unroll
            for (int j = 0; j < EJ; ++j) {
                const unsigned short* p = bPtr + k0 + j * (128 * K_DIM);
                ((uint4*)&bF[j])[0] = *(const uint4*)(p);           // K=k0+8h..
                ((uint4*)&bF[j])[1] = *(const uint4*)(p + 16);      // K=k0+16+8h..
            }

            #pragma unroll
            for (int j = 0; j < EJ; ++j)
                #pragma unroll
                for (int mt = 0; mt < MT; ++mt)
                    acc[mt][j] = __builtin_amdgcn_wmma_f32_16x16x32_bf16(
                        false, aF[mt], false, bF[j], (short)0, acc[mt][j],
                        false, false);
        }

        // Epilogue: + proj_b + gathered inner_pos; streaming (NT) stores.
        #pragma unroll
        for (int j = 0; j < EJ; ++j) {
            const int e  = half * 384 + j * 128 + wave * 16 + nloc;
            const float pb = proj_b[e];
            #pragma unroll
            for (int mt = 0; mt < MT; ++mt) {
                #pragma unroll
                for (int r = 0; r < 8; ++r) {
                    int lr = mt * 16 + r + 8 * h;  // C/D layout: VGPR r -> M=r(+8h)
                    int g  = r0 + lr;
                    float pos = inner_pos[(size_t)s_mid[lr] * E_DIM + e];
                    __builtin_nontemporal_store(acc[mt][j][r] + pb + pos,
                                                out + (size_t)g * E_DIM + e);
                }
            }
        }
    }
}

// ---------------------------------------------------------------------------
// Launch
// ---------------------------------------------------------------------------
extern "C" void kernel_launch(void* const* d_in, const int* in_sizes, int n_in,
                              void* d_out, int out_size, void* d_ws, size_t ws_size,
                              hipStream_t stream) {
    const float* img         = (const float*)d_in[0];
    const float* noise_outer = (const float*)d_in[1];
    const float* noise_inner = (const float*)d_in[2];
    const float* proj_w      = (const float*)d_in[3];
    const float* proj_b      = (const float*)d_in[4];
    const float* inner_pos   = (const float*)d_in[5];
    float* out = (float*)d_out;

    // Workspace layout (needs ~1.29 MB):
    //   [0)        int  ids_keep   [256]
    //   [1024)     int  ids_keep_p [25088]
    //   [101376)   bf16 proj_w     [768*768]
    char* ws = (char*)d_ws;
    int* wkeep            = (int*)(ws);
    int* wkeepp           = (int*)(ws + 1024);
    unsigned short* wbf   = (unsigned short*)(ws + 1024 + ROWS * 4);  // 101376

    float* out_keep    = out + X_SIZE;
    float* out_drop    = out + X_SIZE + BATCH * NKEEP;
    float* out_restore = out + X_SIZE + 2 * BATCH * NKEEP;

    k_outer<<<BATCH, 128, 0, stream>>>(noise_outer, wkeep, out_keep, out_drop, out_restore);
    k_inner<<<BATCH * NKEEP, 256, 0, stream>>>(noise_inner, wkeepp);
    k_wconv<<<(E_DIM * K_DIM) / 1024, 256, 0, stream>>>(proj_w, wbf);
    k_gemm<<<ROWS / 32, 256, 0, stream>>>(img, wkeep, wkeepp, wbf, proj_b, inner_pos, out);
}
